// EffiAttention_64896955842941
// MI455X (gfx1250) — matrix operand order repaired and checked by reading, not compiled, not run on hardware
//
#include <hip/hip_runtime.h>
#include <hip/hip_bf16.h>
#include <stdint.h>

// ---------------------------------------------------------------------------
// Types for CDNA5 WMMA (wave32, 16x16x32 bf16 -> f32 accum)
// ---------------------------------------------------------------------------
typedef __attribute__((ext_vector_type(16))) __bf16 v16bf;
typedef __attribute__((ext_vector_type(8)))  __bf16 v8bf;
typedef __attribute__((ext_vector_type(8)))  float  v8f;

#define B_   4
#define N_   2048
#define C_   1024
#define H_   16
#define DH_  64
#define TSB  72   // LDS row stride in bf16 elems (144B = 9*16B: aligned, conflict-free)

// A-matrix fragment (16xK, 16-bit): lane L<16 -> row M=L, K {0..7,16..23};
// lanes 16-31 -> row M=L-16, K {8..15,24..31}.
__device__ __forceinline__ v16bf frag_a(const __bf16* base, int stride) {
    const int lane = threadIdx.x & 31;
    const __bf16* p = base + (lane & 15) * stride + ((lane >> 4) << 3);
    v8bf lo = *(const v8bf*)(p);
    v8bf hi = *(const v8bf*)(p + 16);
    return __builtin_shufflevector(lo, hi, 0,1,2,3,4,5,6,7,8,9,10,11,12,13,14,15);
}

// B-matrix fragment (Kx16, 16-bit): lane L<16 -> col N=L, K {0..7,16..23};
// lanes 16-31 -> col N=L-16, K {8..15,24..31}. `base` is a [N][K] layout.
__device__ __forceinline__ v16bf frag_b(const __bf16* base, int stride) {
    const int lane = threadIdx.x & 31;
    const __bf16* p = base + (lane & 15) * stride + ((lane >> 4) << 3);
    v8bf lo = *(const v8bf*)(p);
    v8bf hi = *(const v8bf*)(p + 16);
    return __builtin_shufflevector(lo, hi, 0,1,2,3,4,5,6,7,8,9,10,11,12,13,14,15);
}

__device__ __forceinline__ v8f wmma_bf16(v16bf a, v16bf b, v8f c) {
    return __builtin_amdgcn_wmma_f32_16x16x32_bf16(false, a, false, b, (short)0, c,
                                                   false, false);
}

__device__ __forceinline__ v8f vzero8() {
    v8f z;
#pragma unroll
    for (int i = 0; i < 8; ++i) z[i] = 0.0f;
    return z;
}

// CDNA5 async global->LDS copy (16B), tracked by ASYNCcnt.
__device__ __forceinline__ void async_b128(const __bf16* g, __bf16* l) {
    const unsigned lds = (unsigned)(size_t)l;   // low 32 bits = LDS byte offset
    asm volatile("global_load_async_to_lds_b128 %0, %1, off"
                 :: "v"(lds), "v"(g) : "memory");
}
__device__ __forceinline__ void wait_async0() {
    asm volatile("s_wait_asynccnt 0x0" ::: "memory");
}

// Row reductions within 16-lane groups (matches C-matrix row striping).
__device__ __forceinline__ float rowmax16(float v) {
#pragma unroll
    for (int m = 1; m <= 8; m <<= 1) v = fmaxf(v, __shfl_xor(v, m, 32));
    return v;
}
__device__ __forceinline__ float rowsum16(float v) {
#pragma unroll
    for (int m = 1; m <= 8; m <<= 1) v += __shfl_xor(v, m, 32);
    return v;
}

// ---------------------------------------------------------------------------
// One-time prep: f32 -> bf16 convert, and weight transpose+convert to [n][k]
// ---------------------------------------------------------------------------
__global__ __launch_bounds__(256) void convert_bf16_kernel(const float* __restrict__ s,
                                                           __bf16* __restrict__ d) {
    const size_t i = ((size_t)blockIdx.x * 256 + threadIdx.x) * 8;
    const float4 a = *(const float4*)(s + i);
    const float4 b = *(const float4*)(s + i + 4);
    v8bf o;
    o[0] = (__bf16)a.x; o[1] = (__bf16)a.y; o[2] = (__bf16)a.z; o[3] = (__bf16)a.w;
    o[4] = (__bf16)b.x; o[5] = (__bf16)b.y; o[6] = (__bf16)b.z; o[7] = (__bf16)b.w;
    *(v8bf*)(d + i) = o;
}

__global__ __launch_bounds__(256) void transpose_w(const float* __restrict__ w,
                                                   __bf16* __restrict__ wt,
                                                   int K, int Nn) {
    __shared__ __bf16 t[32][33];
    const int nb = blockIdx.x * 32;
    const int kb = blockIdx.y * 32;
    const int tx = threadIdx.x & 31, ty = threadIdx.x >> 5;
#pragma unroll
    for (int i = 0; i < 32; i += 8)
        t[ty + i][tx] = (__bf16)w[(size_t)(kb + ty + i) * Nn + nb + tx];
    __syncthreads();
#pragma unroll
    for (int i = 0; i < 32; i += 8)
        wt[(size_t)(nb + ty + i) * K + kb + tx] = t[tx][ty + i];
}

// ---------------------------------------------------------------------------
// Shared GEMM core: C[128x128] block, 8 waves (4x2), 32x64 per wave,
// K-step 64, async double-buffered LDS, 16 wmma per wave per iteration.
// A: [M][1024] bf16 row-major.  BT: [N][1024] bf16 row-major (B transposed).
// ---------------------------------------------------------------------------
__device__ __forceinline__ void fetch_tiles_async(const __bf16* __restrict__ A,
                                                  const __bf16* __restrict__ BT,
                                                  __bf16* As, __bf16* Bs,
                                                  int mBase, int nBase, int kk, int tid) {
#pragma unroll
    for (int i = 0; i < 4; ++i) {
        const int slot = tid + 256 * i;
        const int row = slot >> 3, u = (slot & 7) * 8;
        async_b128(A + (size_t)(mBase + row) * C_ + kk + u, As + row * TSB + u);
    }
#pragma unroll
    for (int i = 0; i < 4; ++i) {
        const int slot = tid + 256 * i;
        const int row = slot >> 3, u = (slot & 7) * 8;
        async_b128(BT + (size_t)(nBase + row) * C_ + kk + u, Bs + row * TSB + u);
    }
}

__device__ __forceinline__ void compute_tile(const __bf16* As, const __bf16* Bs,
                                             v8f acc[2][4], int wm, int wn) {
#pragma unroll
    for (int ks = 0; ks < 2; ++ks) {
        const v16bf a0 = frag_a(As + (wm * 32 + 0)  * TSB + ks * 32, TSB);
        const v16bf a1 = frag_a(As + (wm * 32 + 16) * TSB + ks * 32, TSB);
#pragma unroll
        for (int j = 0; j < 4; ++j) {
            const v16bf b = frag_b(Bs + (wn * 64 + j * 16) * TSB + ks * 32, TSB);
            acc[0][j] = wmma_bf16(a0, b, acc[0][j]);
            acc[1][j] = wmma_bf16(a1, b, acc[1][j]);
        }
    }
}

__device__ __forceinline__ void gemm_main(const __bf16* A, const __bf16* BT,
                                          __bf16* As0, __bf16* As1,
                                          __bf16* Bs0, __bf16* Bs1,
                                          v8f acc[2][4],
                                          int mBase, int nBase, int tid,
                                          int wm, int wn) {
    fetch_tiles_async(A, BT, As0, Bs0, mBase, nBase, 0, tid);
    wait_async0();
    __syncthreads();
    for (int kk = 0; kk < C_; kk += 64) {
        const int cur = (kk >> 6) & 1;
        __bf16* Asc = cur ? As1 : As0;
        __bf16* Bsc = cur ? Bs1 : Bs0;
        if (kk + 64 < C_)
            fetch_tiles_async(A, BT, cur ? As0 : As1, cur ? Bs0 : Bs1,
                              mBase, nBase, kk + 64, tid);
        compute_tile(Asc, Bsc, acc, wm, wn);
        wait_async0();
        __syncthreads();
    }
}

// ---------------------------------------------------------------------------
// Kernel: qkv = xb @ WqkvT^T, scattered into q[bh][n][d], k[bh][n][d], vt[bh][d][n]
// ---------------------------------------------------------------------------
__global__ __launch_bounds__(256) void qkv_gemm(const __bf16* __restrict__ xb,
                                                const __bf16* __restrict__ wqkvT,
                                                __bf16* __restrict__ q,
                                                __bf16* __restrict__ k,
                                                __bf16* __restrict__ vt) {
    __shared__ __bf16 As[2][128 * TSB];
    __shared__ __bf16 Bs[2][128 * TSB];
    const int tid = threadIdx.x, lane = tid & 31, wave = tid >> 5;
    const int wm = wave >> 1, wn = wave & 1;
    const int mBase = blockIdx.y * 128;
    const int nBase = blockIdx.x * 128;

    v8f acc[2][4];
#pragma unroll
    for (int i = 0; i < 2; ++i)
#pragma unroll
        for (int j = 0; j < 4; ++j) acc[i][j] = vzero8();

    gemm_main(xb, wqkvT, As[0], As[1], Bs[0], Bs[1], acc, mBase, nBase, tid, wm, wn);

    // s (q/k/v select) is uniform per block: 128 | 1024
    const int s = nBase >> 10;
    const int colBase = (nBase & (C_ - 1)) + wn * 64;
    const int rowBase = mBase + wm * 32 + ((lane >> 4) << 3);
    if (s == 2) {
#pragma unroll
        for (int i = 0; i < 2; ++i)
#pragma unroll
            for (int j = 0; j < 4; ++j)
#pragma unroll
                for (int r = 0; r < 8; ++r) {
                    const int row = rowBase + i * 16 + r;
                    const int col = colBase + j * 16 + (lane & 15);
                    const int b = row >> 11, n = row & (N_ - 1);
                    const size_t bh = (size_t)(b * H_ + (col >> 6));
                    vt[(bh * DH_ + (col & 63)) * N_ + n] = (__bf16)acc[i][j][r];
                }
    } else {
        __bf16* dst = (s == 0) ? q : k;
#pragma unroll
        for (int i = 0; i < 2; ++i)
#pragma unroll
            for (int j = 0; j < 4; ++j)
#pragma unroll
                for (int r = 0; r < 8; ++r) {
                    const int row = rowBase + i * 16 + r;
                    const int col = colBase + j * 16 + (lane & 15);
                    const int b = row >> 11, n = row & (N_ - 1);
                    const size_t bh = (size_t)(b * H_ + (col >> 6));
                    dst[(bh * N_ + n) * DH_ + (col & 63)] = (__bf16)acc[i][j][r];
                }
    }
}

// ---------------------------------------------------------------------------
// Flash attention: grid = B*H*(N/64), 128 threads (4 waves), 16 q-rows/wave,
// streaming 64-key tiles: 8 wmma for S, 8 wmma for P@V per tile.
// ---------------------------------------------------------------------------
__global__ __launch_bounds__(128) void flash_attn(const __bf16* __restrict__ q,
                                                  const __bf16* __restrict__ k,
                                                  const __bf16* __restrict__ vt,
                                                  __bf16* __restrict__ out) {
    __shared__ __bf16 Pbuf[4][16 * 64];
    const int tid = threadIdx.x, lane = tid & 31, wave = tid >> 5;
    const int bh = blockIdx.x >> 5;
    const int qt = blockIdx.x & 31;
    const int n0 = qt * 64 + wave * 16;
    const float scale = 0.125f;

    const __bf16* qbase = q + ((size_t)bh * N_ + n0) * DH_;
    const v16bf q0 = frag_a(qbase,      DH_);
    const v16bf q1 = frag_a(qbase + 32, DH_);
    __bf16* PbufW = &Pbuf[wave][0];

    float m[8], l[8];
    v8f o[4];
#pragma unroll
    for (int r = 0; r < 8; ++r) { m[r] = -3.0e38f; l[r] = 0.0f; }
#pragma unroll
    for (int j = 0; j < 4; ++j) o[j] = vzero8();

    for (int kt = 0; kt < N_; kt += 64) {
        const __bf16* kb = k + ((size_t)bh * N_ + kt) * DH_;

        v8f s[4];
#pragma unroll
        for (int t = 0; t < 4; ++t) {
            s[t] = vzero8();
            s[t] = wmma_bf16(q0, frag_b(kb + t * 16 * DH_,      DH_), s[t]);
            s[t] = wmma_bf16(q1, frag_b(kb + t * 16 * DH_ + 32, DH_), s[t]);
        }

        if (kt + 64 < N_) {
            __builtin_prefetch(kb + 64 * DH_ + lane * DH_, 0, 0);
            __builtin_prefetch(vt + ((size_t)bh * DH_ + lane) * N_ + kt + 64, 0, 0);
        }

        // Online softmax; rows live in 16-lane groups of each C VGPR index r
#pragma unroll
        for (int r = 0; r < 8; ++r) {
            float a0 = s[0][r] * scale, a1 = s[1][r] * scale;
            float a2 = s[2][r] * scale, a3 = s[3][r] * scale;
            const float tmax = rowmax16(fmaxf(fmaxf(a0, a1), fmaxf(a2, a3)));
            const float mn = fmaxf(m[r], tmax);
            const float alpha = __expf(m[r] - mn);
            const float p0 = __expf(a0 - mn), p1 = __expf(a1 - mn);
            const float p2 = __expf(a2 - mn), p3 = __expf(a3 - mn);
            l[r] = l[r] * alpha + rowsum16((p0 + p1) + (p2 + p3));
            m[r] = mn;
#pragma unroll
            for (int j = 0; j < 4; ++j) o[j][r] *= alpha;
            const int prow = r + ((lane >> 4) << 3);
            const int pcol = lane & 15;
            PbufW[prow * 64 + pcol +  0] = (__bf16)p0;
            PbufW[prow * 64 + pcol + 16] = (__bf16)p1;
            PbufW[prow * 64 + pcol + 32] = (__bf16)p2;
            PbufW[prow * 64 + pcol + 48] = (__bf16)p3;
        }

        // O += P @ V  (K-depth 64 over this key tile)
        const v16bf pf0 = frag_a(PbufW,      64);
        const v16bf pf1 = frag_a(PbufW + 32, 64);
#pragma unroll
        for (int j = 0; j < 4; ++j) {
            const __bf16* vb = vt + ((size_t)bh * DH_ + j * 16) * N_ + kt;
            o[j] = wmma_bf16(pf0, frag_b(vb,      N_), o[j]);
            o[j] = wmma_bf16(pf1, frag_b(vb + 32, N_), o[j]);
        }
    }

    const int b = bh >> 4, h = bh & 15;
#pragma unroll
    for (int r = 0; r < 8; ++r) {
        const float inv = 1.0f / l[r];
        const int n = n0 + r + ((lane >> 4) << 3);
#pragma unroll
        for (int j = 0; j < 4; ++j) {
            const int col = h * DH_ + j * 16 + (lane & 15);
            out[((size_t)b * N_ + n) * C_ + col] = (__bf16)(o[j][r] * inv);
        }
    }
}

// ---------------------------------------------------------------------------
// Kernel: final = attn @ WprojT^T + bproj  (f32 out)
// ---------------------------------------------------------------------------
__global__ __launch_bounds__(256) void proj_gemm(const __bf16* __restrict__ a,
                                                 const __bf16* __restrict__ wprojT,
                                                 const float* __restrict__ bias,
                                                 float* __restrict__ out) {
    __shared__ __bf16 As[2][128 * TSB];
    __shared__ __bf16 Bs[2][128 * TSB];
    const int tid = threadIdx.x, lane = tid & 31, wave = tid >> 5;
    const int wm = wave >> 1, wn = wave & 1;
    const int mBase = blockIdx.y * 128;
    const int nBase = blockIdx.x * 128;

    v8f acc[2][4];
#pragma unroll
    for (int i = 0; i < 2; ++i)
#pragma unroll
        for (int j = 0; j < 4; ++j) acc[i][j] = vzero8();

    gemm_main(a, wprojT, As[0], As[1], Bs[0], Bs[1], acc, mBase, nBase, tid, wm, wn);

#pragma unroll
    for (int j = 0; j < 4; ++j) {
        const int col = nBase + wn * 64 + j * 16 + (lane & 15);
        const float bb = bias[col];
#pragma unroll
        for (int i = 0; i < 2; ++i)
#pragma unroll
            for (int r = 0; r < 8; ++r) {
                const int row = mBase + wm * 32 + i * 16 + r + ((lane >> 4) << 3);
                out[(size_t)row * C_ + col] = acc[i][j][r] + bb;
            }
    }
}

// ---------------------------------------------------------------------------
// Launch
// ---------------------------------------------------------------------------
extern "C" void kernel_launch(void* const* d_in, const int* in_sizes, int n_in,
                              void* d_out, int out_size, void* d_ws, size_t ws_size,
                              hipStream_t stream) {
    const float* x     = (const float*)d_in[0];
    // d_in[1] = xpos: unused by the reference computation
    const float* Wqkv  = (const float*)d_in[2];
    const float* Wproj = (const float*)d_in[3];
    const float* bproj = (const float*)d_in[4];
    float* out = (float*)d_out;

    const size_t XS = (size_t)B_ * N_ * C_;        // 8,388,608
    __bf16* xb     = (__bf16*)d_ws;                 // also reused as attn output
    __bf16* wqkvT  = xb + XS;                       // [3072][1024]
    __bf16* wprojT = wqkvT + (size_t)3 * C_ * C_;   // [1024][1024]
    __bf16* qws    = wprojT + (size_t)C_ * C_;
    __bf16* kws    = qws + XS;
    __bf16* vtws   = kws + XS;                      // V transposed: [bh][d][n]
    __bf16* attn   = xb;                            // reuse after qkv_gemm is done

    convert_bf16_kernel<<<dim3((unsigned)(XS / (256 * 8))), 256, 0, stream>>>(x, xb);
    transpose_w<<<dim3((3 * C_) / 32, C_ / 32), 256, 0, stream>>>(Wqkv, wqkvT, C_, 3 * C_);
    transpose_w<<<dim3(C_ / 32, C_ / 32), 256, 0, stream>>>(Wproj, wprojT, C_, C_);

    qkv_gemm<<<dim3((3 * C_) / 128, (B_ * N_) / 128), 256, 0, stream>>>(
        xb, wqkvT, qws, kws, vtws);
    flash_attn<<<dim3(B_ * H_ * (N_ / 64)), 128, 0, stream>>>(
        qws, kws, vtws, attn);
    proj_gemm<<<dim3(C_ / 128, (B_ * N_) / 128), 256, 0, stream>>>(
        attn, wprojT, bproj, out);
}
